// ModMultiHeadAttention_67130338836869
// MI455X (gfx1250) — matrix operand order repaired and checked by reading, not compiled, not run on hardware
//
#include <hip/hip_runtime.h>
#include <hip/hip_bf16.h>

// MI455X (gfx1250, wave32) fused MHA:
//   k = x@Wk+bk ; v = x@Wv+bv ; flash-attn(q=x, k, v, causal) ; out = attn@Wo+bo
// All matmuls on v_wmma_f32_16x16x32_f16 (f16 in, f32 accumulate).
// GEMMs stage tiles into LDS via GLOBAL_LOAD_ASYNC_TO_LDS_B128 (ASYNCcnt),
// double-buffered so the next tile's DMA overlaps the current tile's WMMAs.
//
// Workspace (<= ~34 MiB):
//   [ 0MiB) xh   f16 [B*S, D]
//   [ 8MiB) Kh   f16 [B*S, D]
//   [16MiB) VTh  f16 [B, D, S]   (transposed V for contiguous P*V B-fragments)
//   [24MiB) Ah   f16 [B*S, D]    (attention output, pre-projection)
//   [32MiB) WkT/WvT/WoT f16 [N, K] each (pre-transposed weights)

typedef __attribute__((ext_vector_type(16))) _Float16 v16h;
typedef __attribute__((ext_vector_type(8)))  _Float16 v8h;
typedef __attribute__((ext_vector_type(8)))  float    v8f;
typedef int v4i __attribute__((vector_size(16)));  // matches builtin signature

union V16 { v16h v; v8h h[2]; };

#define AS1 __attribute__((address_space(1)))
#define AS3 __attribute__((address_space(3)))

#if __has_builtin(__builtin_amdgcn_global_load_async_to_lds_b128)
#define HAS_ASYNC_LDS 1
#else
#define HAS_ASYNC_LDS 0
#endif

__device__ __forceinline__ void async_cp16(const _Float16* g, _Float16* l) {
#if HAS_ASYNC_LDS
  // 16B per lane, global -> LDS, tracked by ASYNCcnt (no VGPR data path)
  __builtin_amdgcn_global_load_async_to_lds_b128((AS1 v4i*)g, (AS3 v4i*)l, 0, 0);
#else
  *(v8h*)l = *(const v8h*)g;  // sync fallback: b128 load + ds_store_b128
#endif
}

__device__ __forceinline__ void wait_async0() {
#if HAS_ASYNC_LDS
#if __has_builtin(__builtin_amdgcn_s_wait_asynccnt)
  __builtin_amdgcn_s_wait_asynccnt(0);
#else
  asm volatile("s_wait_asynccnt 0x0" ::: "memory");
#endif
#endif
}

__device__ __forceinline__ v8f wmma_f16(v16h a, v16h b, v8f c) {
  // 8 args: (neg_a, A, neg_b, B, c_mod, C, reuse_a, reuse_b)
  return __builtin_amdgcn_wmma_f32_16x16x32_f16(false, a, false, b, (short)0, c,
                                                false, false);
}

__device__ __forceinline__ float redmax16(float v) {
#pragma unroll
  for (int m = 1; m < 16; m <<= 1) v = fmaxf(v, __shfl_xor(v, m, 32));
  return v;
}
__device__ __forceinline__ float redsum16(float v) {
#pragma unroll
  for (int m = 1; m < 16; m <<= 1) v += __shfl_xor(v, m, 32);
  return v;
}

// ---------------- conversion kernels ----------------

__global__ void f32_to_f16_kernel(const float* __restrict__ in,
                                  _Float16* __restrict__ out, int n) {
  int i = blockIdx.x * blockDim.x + threadIdx.x;
  if (i < n) out[i] = (_Float16)in[i];
}

// W [512,512] row-major (K x N)  ->  Wt [N][K] f16
__global__ void w_transpose_f16_kernel(const float* __restrict__ W,
                                       _Float16* __restrict__ Wt) {
  int t = blockIdx.x * blockDim.x + threadIdx.x;  // 262144 threads
  int k = t >> 9, n = t & 511;
  Wt[n * 512 + k] = (_Float16)W[t];
}

// -------- WMMA GEMM with async-LDS double buffering --------
// Block = 4 waves sharing one 16-row A tile; block covers 256 N columns
// (one 16x64 tile per wave). Per 32-wide K step: stage A(16x32)+W(256x32)
// into LDS asynchronously for step s+1 while running 4 WMMAs on step s.
// mode: 0 -> f16 [M,N]; 1 -> f16 transposed [B,D,S]; 2 -> f32 [M,N].

__launch_bounds__(128)
__global__ void gemm_wmma_async(const _Float16* __restrict__ A,
                                const _Float16* __restrict__ Wt,
                                const float* __restrict__ bias,
                                _Float16* __restrict__ outH,
                                float* __restrict__ outF,
                                int mode) {
  // per buffer: A tile 512 halves (1KB) + W tile 8192 halves (16KB)
  __shared__ __align__(16) _Float16 sm[2 * 8704];

  const int tid  = threadIdx.x;
  const int w    = tid >> 5;
  const int lane = tid & 31;
  const int lid  = lane & 15;
  const int half = lane >> 4;
  const int m0   = (blockIdx.x >> 1) << 4;  // 512 M tiles
  const int nb   = (blockIdx.x & 1) << 8;   // 0 or 256
  const int n0   = nb + (w << 6);

  v8f acc[4];
#pragma unroll
  for (int c = 0; c < 4; ++c) {
    float bvv = bias[n0 + c * 16 + lid];
#pragma unroll
    for (int e = 0; e < 8; ++e) acc[c][e] = bvv;
  }

  auto stage = [&](int k0, int buf) {
    _Float16* base = sm + buf * 8704;
    if (tid < 64) {  // A tile: 16 rows x 32 halves = 64 x 16B chunks
      int row = tid >> 2, c = tid & 3;
      async_cp16(A + (m0 + row) * 512 + k0 + c * 8, base + tid * 8);
    }
    _Float16* wb = base + 512;  // W tile: 256 rows x 32 halves = 1024 chunks
#pragma unroll
    for (int i = 0; i < 8; ++i) {
      int r = tid + i * 128;
      int nl = r >> 2, c = r & 3;
      async_cp16(Wt + (nb + nl) * 512 + k0 + c * 8, wb + r * 8);
    }
  };

  stage(0, 0);
  for (int s = 0; s < 16; ++s) {
    wait_async0();      // this wave's portion of stage(s) landed in LDS
    __syncthreads();    // everyone's portion landed
    if (s + 1 < 16) stage((s + 1) * 32, (s + 1) & 1);  // overlap with WMMAs

    _Float16* base = sm + (s & 1) * 8704;
    V16 a;  // A-fragment: lanes<16 hold K {0..15}, lanes>=16 {16..31}
    const _Float16* ar = base + lid * 32;
    a.h[0] = *(const v8h*)(ar + half * 16);
    a.h[1] = *(const v8h*)(ar + half * 16 + 8);
    const _Float16* wbb = base + 512;
#pragma unroll
    for (int c = 0; c < 4; ++c) {
      v16h bw = *(const v16h*)(wbb + (w * 64 + c * 16 + lid) * 32 + half * 16);
      acc[c] = wmma_f16(a.v, bw, acc[c]);
    }
  }

  if (mode == 0) {
#pragma unroll
    for (int c = 0; c < 4; ++c)
#pragma unroll
      for (int r = 0; r < 8; ++r)
        outH[(m0 + r + 8 * half) * 512 + n0 + c * 16 + lid] = (_Float16)acc[c][r];
  } else if (mode == 1) {
    // transposed store [B, D, S]: per lane 8 consecutive f16 along S -> b128 store
    const int b  = m0 >> 11;  // S = 2048
    const int s0 = (m0 & 2047) + half * 8;
#pragma unroll
    for (int c = 0; c < 4; ++c) {
      v8h hv;
#pragma unroll
      for (int r = 0; r < 8; ++r) hv[r] = (_Float16)acc[c][r];
      *(v8h*)(outH + ((b * 512 + n0 + c * 16 + lid) << 11) + s0) = hv;
    }
  } else {
#pragma unroll
    for (int c = 0; c < 4; ++c)
#pragma unroll
      for (int r = 0; r < 8; ++r)
        outF[(m0 + r + 8 * half) * 512 + n0 + c * 16 + lid] = acc[c][r];
  }
}

// ---------------- causal flash-attention, one wave = 16 query rows ----------
// Block = 4 waves (64 query rows of one (b,h)). Grid = B*H*(S/64) = 1024.
// Per 32-key tile: 4 score WMMAs (hd=64), online softmax (16-lane shuffles),
// P bounced via per-wave LDS (C-layout f32 -> A-layout f16), 4 P*V WMMAs.

__launch_bounds__(128)
__global__ void attn_wmma(const _Float16* __restrict__ Xh,
                          const _Float16* __restrict__ Kh,
                          const _Float16* __restrict__ VTh,
                          _Float16* __restrict__ Oh) {
  __shared__ __align__(16) unsigned char smem[4 * 4096];  // 4 KiB per wave

  const int qt   = blockIdx.x & 31;  // S/64 tiles
  const int bh   = blockIdx.x >> 5;
  const int b    = bh >> 3;
  const int h    = bh & 7;
  const int w    = threadIdx.x >> 5;
  const int lane = threadIdx.x & 31;
  const int lid  = lane & 15;
  const int half = lane >> 4;
  const int q0   = qt * 64 + w * 16;

  const _Float16* qbase = Xh + (b * 2048) * 512 + h * 64;
  const _Float16* kbase = Kh + (b * 2048) * 512 + h * 64;
  const _Float16* vbase = VTh + (b * 512 + h * 64) * 2048;

  _Float16* pbuf = (_Float16*)(smem + w * 4096);  // 16x32 f16 (in loop)
  float*    obuf = (float*)(smem + w * 4096);     // 16x64 f32 (after loop)

  // Q A-fragments for both 32-wide d chunks (hd = 64), loaded once.
  V16 aq[2];
  {
    const _Float16* qr = qbase + (q0 + lid) * 512;
#pragma unroll
    for (int dc = 0; dc < 2; ++dc) {
      aq[dc].h[0] = *(const v8h*)(qr + dc * 32 + half * 16);
      aq[dc].h[1] = *(const v8h*)(qr + dc * 32 + half * 16 + 8);
    }
  }

  v8f oacc[4];
#pragma unroll
  for (int c = 0; c < 4; ++c)
#pragma unroll
    for (int e = 0; e < 8; ++e) oacc[c][e] = 0.0f;
  float rm[8], rl[8];
#pragma unroll
  for (int r = 0; r < 8; ++r) { rm[r] = -3.0e38f; rl[r] = 0.0f; }

  const int nkt = (q0 + 47) >> 5;  // causal key-tile count (32 keys per tile)
  for (int t = 0; t < nkt; ++t) {
    const int kb = t * 32;

    // scores: S(16x32) = Q(16x64) . K(32x64)^T, scaled by 1/sqrt(64)
    v8f sc[2];
#pragma unroll
    for (int g = 0; g < 2; ++g) {
      v8f s;
#pragma unroll
      for (int e = 0; e < 8; ++e) s[e] = 0.0f;
      const _Float16* kr = kbase + (kb + g * 16 + lid) * 512;
#pragma unroll
      for (int dc = 0; dc < 2; ++dc) {
        v16h bk = *(const v16h*)(kr + dc * 32 + half * 16);
        s = wmma_f16(aq[dc].v, bk, s);
      }
      sc[g] = s * 0.125f;
    }

    if (kb + 31 > q0) {  // boundary tile(s): causal mask
#pragma unroll
      for (int g = 0; g < 2; ++g) {
        const int key = kb + g * 16 + lid;
#pragma unroll
        for (int r = 0; r < 8; ++r)
          if (key > q0 + r + 8 * half) sc[g][r] = -1.0e9f;
      }
    }

    // online softmax per row (row = C-layout VGPR r within a 16-lane half)
#pragma unroll
    for (int r = 0; r < 8; ++r) {
      float tmax = redmax16(fmaxf(sc[0][r], sc[1][r]));
      float mnew = fmaxf(rm[r], tmax);
      float scl  = __expf(rm[r] - mnew);
      rm[r] = mnew;
      float p0 = __expf(sc[0][r] - mnew);
      float p1 = __expf(sc[1][r] - mnew);
      sc[0][r] = p0; sc[1][r] = p1;
      rl[r] = rl[r] * scl + redsum16(p0 + p1);
#pragma unroll
      for (int c = 0; c < 4; ++c) oacc[c][r] *= scl;
    }

    // P: C-layout f32 -> LDS -> A-layout f16 (per-wave region, DS is in-order)
#pragma unroll
    for (int g = 0; g < 2; ++g)
#pragma unroll
      for (int r = 0; r < 8; ++r)
        pbuf[(r + 8 * half) * 32 + g * 16 + lid] = (_Float16)sc[g][r];
    asm volatile("" ::: "memory");
    V16 ap;
    {
      const _Float16* pr = pbuf + lid * 32;
      ap.h[0] = *(const v8h*)(pr + half * 16);
      ap.h[1] = *(const v8h*)(pr + half * 16 + 8);
    }

    // O(16x64) += P(16x32) . V(32x64); VTh rows are d, contiguous along keys
#pragma unroll
    for (int c = 0; c < 4; ++c) {
      v16h bv = *(const v16h*)(vbase + ((c * 16 + lid) << 11) + kb + half * 16);
      oacc[c] = wmma_f16(ap.v, bv, oacc[c]);
    }
  }

  // normalize and bounce through LDS for coalesced f16 stores
#pragma unroll
  for (int r = 0; r < 8; ++r) rl[r] = 1.0f / rl[r];
#pragma unroll
  for (int c = 0; c < 4; ++c)
#pragma unroll
    for (int r = 0; r < 8; ++r)
      obuf[(r + 8 * half) * 64 + c * 16 + lid] = oacc[c][r] * rl[r];
  asm volatile("" ::: "memory");

  const int row = lane >> 1;
  const int ch  = (lane & 1) * 32;
  const float* src = obuf + row * 64 + ch;
  _Float16* dst = Oh + (b * 2048 + q0 + row) * 512 + h * 64 + ch;
#pragma unroll
  for (int j = 0; j < 32; j += 8) {
    v8h hv;
#pragma unroll
    for (int e = 0; e < 8; ++e) hv[e] = (_Float16)src[j + e];
    *(v8h*)(dst + j) = hv;
  }
}

// ---------------- launcher ----------------

extern "C" void kernel_launch(void* const* d_in, const int* in_sizes, int n_in,
                              void* d_out, int out_size, void* d_ws, size_t ws_size,
                              hipStream_t stream) {
  const float* x  = (const float*)d_in[0];
  const float* Wk = (const float*)d_in[1];
  const float* bk = (const float*)d_in[2];
  const float* Wv = (const float*)d_in[3];
  const float* bv = (const float*)d_in[4];
  const float* Wo = (const float*)d_in[5];
  const float* bo = (const float*)d_in[6];
  float* out = (float*)d_out;

  char* ws = (char*)d_ws;
  const size_t MB = 1024u * 1024u;
  _Float16* xh  = (_Float16*)(ws);
  _Float16* Kh  = (_Float16*)(ws + 8 * MB);
  _Float16* VTh = (_Float16*)(ws + 16 * MB);
  _Float16* Ah  = (_Float16*)(ws + 24 * MB);
  _Float16* WkT = (_Float16*)(ws + 32 * MB);
  _Float16* WvT = (_Float16*)(ws + 32 * MB + 512 * 512 * 2);
  _Float16* WoT = (_Float16*)(ws + 32 * MB + 2 * 512 * 512 * 2);

  const int NX = 4 * 2048 * 512;  // 4,194,304
  f32_to_f16_kernel<<<NX / 256, 256, 0, stream>>>(x, xh, NX);
  w_transpose_f16_kernel<<<1024, 256, 0, stream>>>(Wk, WkT);
  w_transpose_f16_kernel<<<1024, 256, 0, stream>>>(Wv, WvT);
  w_transpose_f16_kernel<<<1024, 256, 0, stream>>>(Wo, WoT);

  // K = x@Wk + bk  (f16 row-major)
  gemm_wmma_async<<<1024, 128, 0, stream>>>(xh, WkT, bk, Kh, nullptr, 0);
  // V^T = (x@Wv + bv)^T  (f16 [B,D,S])
  gemm_wmma_async<<<1024, 128, 0, stream>>>(xh, WvT, bv, VTh, nullptr, 1);
  // causal flash attention (q = x, unprojected)
  attn_wmma<<<1024, 128, 0, stream>>>(xh, Kh, VTh, Ah);
  // out = attn@Wo + bo  (f32 to d_out)
  gemm_wmma_async<<<1024, 128, 0, stream>>>(Ah, WoT, bo, nullptr, out, 2);
}
